// MultiHeadFlashAttention_88124138979463
// MI455X (gfx1250) — matrix operand order repaired and checked by reading, not compiled, not run on hardware
//
#include <hip/hip_runtime.h>
#include <hip/hip_bf16.h>

typedef _Float16 half_t;
typedef __attribute__((ext_vector_type(16))) _Float16 v16h;
typedef __attribute__((ext_vector_type(8)))  _Float16 v8h;
typedef __attribute__((ext_vector_type(4)))  _Float16 v4h;
typedef __attribute__((ext_vector_type(8)))  float    v8f;
typedef __attribute__((ext_vector_type(4)))  float    v4f;

#define EMB    1024
#define HEADS  16
#define HD     64
#define BLK    128
#define BT     16384    // B*T
#define NQKV   3072

__device__ __forceinline__ v8f wmma16(v16h a, v16h b, v8f c) {
  // D = A*B + C, f16 inputs, f32 accumulate
  return __builtin_amdgcn_wmma_f32_16x16x32_f16(false, a, false, b, (short)0, c,
                                                false, false);
}

// A fragment (16x32 f16), LDS row-major [row][k], strideH halfs per row.
// lane<16: row=lane, K{0..7} then K{16..23}; lane>=16: row=lane-16, K{8..15},{24..31}
__device__ __forceinline__ v16h ld_afrag(const half_t* base, int strideH, int lane) {
  const half_t* p = base + (lane & 15) * strideH + ((lane >> 4) << 3);
  union { v16h v; v8h h2[2]; } u;
  u.h2[0] = *(const v8h*)p;
  u.h2[1] = *(const v8h*)(p + 16);
  return u.v;
}

// B fragment (32x16 f16) read from B^T layout [n][k].
// lane<16: col n=lane, K{0..7} then K{16..23}; lane>=16: col n=lane-16, K{8..15},{24..31}
__device__ __forceinline__ v16h ld_bfrag(const half_t* baseT, int strideH, int lane) {
  const half_t* p = baseT + (lane & 15) * strideH + ((lane >> 4) << 3);
  union { v16h v; v8h h2[2]; } u;
  u.h2[0] = *(const v8h*)p;
  u.h2[1] = *(const v8h*)(p + 16);
  return u.v;
}

// Low 32 bits of a generic pointer to __shared__ == group-segment byte offset.
__device__ __forceinline__ uint32_t lds_addr32(const void* p) {
  return (uint32_t)(uintptr_t)p;
}

// CDNA5 async copy: 16B/lane global -> LDS, tracked by ASYNCcnt (GVS mode).
__device__ __forceinline__ void async_copy_b128(uint32_t lds_byte_addr,
                                                uint32_t glb_byte_off,
                                                const void* sbase) {
  asm volatile("global_load_async_to_lds_b128 %0, %1, %2"
               :: "v"(lds_byte_addr), "v"(glb_byte_off), "s"(sbase)
               : "memory");
}

__device__ __forceinline__ void wait_asynccnt0() {
  asm volatile("s_wait_asynccnt 0" ::: "memory");
}

// ---------------------------------------------------------------------------
// Tiled WMMA GEMM: C[M,N] = A[M,K] * Bw[K,N] (+ bias). 128x128 tile, K-step 32.
// 256 threads = 8 waves in 2(M) x 4(N); each wave: 4x2 16x16 accumulators.
// Register double buffering: tile k+1 is fetched from HBM while tile k computes.
// ---------------------------------------------------------------------------
template <typename AT, typename OT, bool BIAS>
__global__ __launch_bounds__(256) void wmma_gemm(const AT* __restrict__ A,
                                                 const float* __restrict__ Bw,
                                                 const float* __restrict__ bias,
                                                 OT* __restrict__ Co,
                                                 int M, int N, int K) {
  __shared__ half_t As[128 * 40];   // [m][k], pad to 40 halfs (80B, 16B aligned)
  __shared__ half_t Bs[128 * 40];   // B^T: [n][k]

  const int tid  = threadIdx.x;
  const int lane = tid & 31;
  const int wid  = tid >> 5;
  const int wm   = (wid >> 2) * 64;   // 0 or 64
  const int wn   = (wid & 3) * 32;    // 0..96
  const int m0   = blockIdx.y * 128;
  const int n0   = blockIdx.x * 128;

  v8f zero = {};
  v8f acc[4][2];
#pragma unroll
  for (int i = 0; i < 4; ++i)
#pragma unroll
    for (int j = 0; j < 2; ++j) acc[i][j] = zero;

  const int am = tid >> 3;           // 0..31 (A tile row per pass)
  const int ak = (tid & 7) << 2;     // 0,4,...,28
  const int bn = tid & 127;          // B^T row (= output col)
  const int bk = (tid >> 7) << 4;    // 0 or 16

  v4h   aregs[4];
  half_t bregs[16];

  auto fetchA = [&](int k0) {
#pragma unroll
    for (int p = 0; p < 4; ++p) {
      const int m = am + p * 32;
      const AT* src = A + (size_t)(m0 + m) * K + (k0 + ak);
      if constexpr (sizeof(AT) == 4) {
        const v4f f = *(const v4f*)(const void*)src;
        aregs[p][0] = (_Float16)f[0]; aregs[p][1] = (_Float16)f[1];
        aregs[p][2] = (_Float16)f[2]; aregs[p][3] = (_Float16)f[3];
      } else {
        aregs[p] = *(const v4h*)(const void*)src;
      }
    }
  };
  auto fetchB = [&](int k0) {
    const float* src = Bw + (size_t)(k0 + bk) * N + (n0 + bn);
#pragma unroll
    for (int i = 0; i < 16; ++i) bregs[i] = (half_t)src[(size_t)i * N];
  };
  auto stageLDS = [&]() {
#pragma unroll
    for (int p = 0; p < 4; ++p)
      *(v4h*)&As[(am + p * 32) * 40 + ak] = aregs[p];
#pragma unroll
    for (int i = 0; i < 2; ++i)
      *(v8h*)&Bs[bn * 40 + bk + i * 8] = *(const v8h*)&bregs[i * 8];
  };

  fetchA(0);
  fetchB(0);
  for (int k0 = 0; k0 < K; k0 += 32) {
    stageLDS();
    __syncthreads();
    if (k0 + 32 < K) {          // overlap next HBM fetch with WMMA compute
      fetchA(k0 + 32);
      fetchB(k0 + 32);
    }
    v16h bfr[2];
#pragma unroll
    for (int nt = 0; nt < 2; ++nt)
      bfr[nt] = ld_bfrag(&Bs[(wn + nt * 16) * 40], 40, lane);
#pragma unroll
    for (int mt = 0; mt < 4; ++mt) {
      v16h a = ld_afrag(&As[(wm + mt * 16) * 40], 40, lane);
#pragma unroll
      for (int nt = 0; nt < 2; ++nt)
        acc[mt][nt] = wmma16(a, bfr[nt], acc[mt][nt]);
    }
    __syncthreads();
  }

  // ---- store: C layout -> VGPR i holds row (i + 8*(lane>>4)), col = lane&15 ----
  const int rhalf = (lane >> 4) << 3;
  const int cl    = lane & 15;
#pragma unroll
  for (int nt = 0; nt < 2; ++nt) {
    const int col = n0 + wn + nt * 16 + cl;
    float bv = 0.0f;
    if constexpr (BIAS) bv = bias[col];
#pragma unroll
    for (int mt = 0; mt < 4; ++mt) {
#pragma unroll
      for (int i = 0; i < 8; ++i) {
        const int row = m0 + wm + mt * 16 + i + rhalf;
        OT* dst = Co + (size_t)row * N + col;
        if constexpr (sizeof(OT) == 4)
          *dst = acc[mt][nt][i] + bv;
        else
          *dst = (OT)acc[mt][nt][i];
      }
    }
  }
}

// ---------------------------------------------------------------------------
// Block-local causal attention for one (batch, block, head).
// Q,K staged with CDNA5 async global->LDS DMA (no VGPR round trip); V staged
// transposed via VGPRs. Scores via WMMA, masked softmax, P round-trips LDS
// to become an A fragment, then P@V via WMMA.
// ---------------------------------------------------------------------------
__global__ __launch_bounds__(256) void attn_kernel(const half_t* __restrict__ qkv,
                                                   half_t* __restrict__ out) {
  __shared__ half_t Qs[128 * 72];       // 18432 B
  __shared__ half_t Ks[128 * 72];       // 18432 B
  __shared__ half_t Vt[64 * 136];       // 17408 B  (V^T: [d][j])
  __shared__ half_t Ps[8][16 * 40];     // 10240 B  (per-wave P staging)

  const int tid  = threadIdx.x;
  const int lane = tid & 31;
  const int wid  = tid >> 5;
  const int bid  = blockIdx.x;          // b*512 + blk*16 + h
  const int h    = bid & 15;
  const int blk  = (bid >> 4) & 31;
  const int b    = bid >> 9;
  const size_t trow = (size_t)b * 4096 + (size_t)blk * BLK;

  // ---- stage Q, K via async-to-LDS; V transposed via VGPRs ----
  {
    const int j  = tid >> 1;           // row 0..127
    const int dh = (tid & 1) * 32;     // half offset within the 64-wide head
    const uint32_t gq = (uint32_t)(((trow + j) * (size_t)NQKV + h * HD + dh) * 2);
#pragma unroll
    for (int c = 0; c < 4; ++c) {
      async_copy_b128(lds_addr32(&Qs[j * 72 + dh + c * 8]), gq + c * 16, qkv);
      async_copy_b128(lds_addr32(&Ks[j * 72 + dh + c * 8]),
                      gq + (uint32_t)(EMB * 2) + c * 16, qkv);
    }
    const half_t* vbase = qkv + (trow + j) * (size_t)NQKV + 2 * EMB + h * HD + dh;
    union { v8h v[4]; half_t e[32]; } vv;
#pragma unroll
    for (int c = 0; c < 4; ++c) vv.v[c] = *(const v8h*)(vbase + c * 8);
#pragma unroll
    for (int i = 0; i < 32; ++i) Vt[(dh + i) * 136 + j] = vv.e[i];
  }
  wait_asynccnt0();                     // my async LDS writes are resident
  __syncthreads();                      // ...and so is everyone else's

  // ---- scores: wave rows [wid*16, wid*16+16) x 128 cols, K=64 ----
  v8f z = {};
  v8f s[8];
#pragma unroll
  for (int jt = 0; jt < 8; ++jt) s[jt] = z;
#pragma unroll
  for (int kc = 0; kc < 64; kc += 32) {
    v16h aq = ld_afrag(&Qs[wid * 16 * 72 + kc], 72, lane);
#pragma unroll
    for (int jt = 0; jt < 8; ++jt) {
      v16h bk = ld_bfrag(&Ks[jt * 16 * 72 + kc], 72, lane);
      s[jt] = wmma16(aq, bk, s[jt]);
    }
  }

  // ---- masked, stable softmax (scale 1/sqrt(64)=0.125; div by sum+1e-6) ----
  const int rhalf = (lane >> 4) << 3;   // 0 or 8
  const int cl    = lane & 15;
  float mx[8], sm[8];
#pragma unroll
  for (int i = 0; i < 8; ++i) {
    const int row = wid * 16 + i + rhalf;
    float m = -3.0e38f;
#pragma unroll
    for (int jt = 0; jt < 8; ++jt) {
      const int col = jt * 16 + cl;
      const float v = s[jt][i] * 0.125f;
      if (col <= row && v > m) m = v;
    }
    mx[i] = m;
  }
#pragma unroll
  for (int i = 0; i < 8; ++i) {
#pragma unroll
    for (int off = 1; off <= 8; off <<= 1) {
      const float o = __shfl_xor(mx[i], off, 32);
      mx[i] = mx[i] > o ? mx[i] : o;
    }
  }
#pragma unroll
  for (int i = 0; i < 8; ++i) {
    const int row = wid * 16 + i + rhalf;
    float acc = 0.0f;
#pragma unroll
    for (int jt = 0; jt < 8; ++jt) {
      const int col = jt * 16 + cl;
      const float v = s[jt][i] * 0.125f;
      const float e = (col <= row) ? __expf(v - mx[i]) : 0.0f;
      s[jt][i] = e;
      acc += e;
    }
    sm[i] = acc;
  }
#pragma unroll
  for (int i = 0; i < 8; ++i) {
#pragma unroll
    for (int off = 1; off <= 8; off <<= 1) sm[i] += __shfl_xor(sm[i], off, 32);
    const float inv = 1.0f / (sm[i] + 1e-6f);
#pragma unroll
    for (int jt = 0; jt < 8; ++jt) s[jt][i] *= inv;
  }

  // ---- O = P @ V: K=128 in 4 chunks of 32; P via per-wave LDS staging ----
  v8f o[4];
#pragma unroll
  for (int nt = 0; nt < 4; ++nt) o[nt] = z;
  half_t* pw = &Ps[wid][0];
#pragma unroll
  for (int ks = 0; ks < 4; ++ks) {
#pragma unroll
    for (int jj = 0; jj < 2; ++jj) {
      const int jt = ks * 2 + jj;
#pragma unroll
      for (int i = 0; i < 8; ++i)
        pw[(i + rhalf) * 40 + jj * 16 + cl] = (half_t)s[jt][i];
    }
    v16h ap = ld_afrag(pw, 40, lane);   // LDS ops are in-order within a wave
#pragma unroll
    for (int nt = 0; nt < 4; ++nt) {
      v16h bv = ld_bfrag(&Vt[nt * 16 * 136 + ks * 32], 136, lane);
      o[nt] = wmma16(ap, bv, o[nt]);
    }
  }

  // ---- write attention output (f16) into [BT][EMB] workspace ----
#pragma unroll
  for (int nt = 0; nt < 4; ++nt) {
    const int col = h * HD + nt * 16 + cl;
#pragma unroll
    for (int i = 0; i < 8; ++i) {
      const int row = wid * 16 + i + rhalf;
      out[(trow + row) * (size_t)EMB + col] = (half_t)o[nt][i];
    }
  }
}

// ---------------------------------------------------------------------------
extern "C" void kernel_launch(void* const* d_in, const int* in_sizes, int n_in,
                              void* d_out, int out_size, void* d_ws, size_t ws_size,
                              hipStream_t stream) {
  (void)in_sizes; (void)n_in; (void)out_size; (void)ws_size;
  const float* x     = (const float*)d_in[0];   // [4,4096,1024]
  const float* Wqkv  = (const float*)d_in[1];   // [1024,3072]
  const float* Wproj = (const float*)d_in[2];   // [1024,1024]
  const float* bproj = (const float*)d_in[3];   // [1024]
  float* out = (float*)d_out;                   // [4,4096,1024]

  half_t* qkv  = (half_t*)d_ws;                       // BT x 3072 f16 (96 MB)
  half_t* attn = qkv + (size_t)BT * NQKV;             // BT x 1024 f16 (32 MB)

  dim3 blk(256);
  // 1) qkv = x @ W_qkv   (f32 in, f16 out)
  wmma_gemm<float, half_t, false>
      <<<dim3(NQKV / 128, BT / 128), blk, 0, stream>>>(x, Wqkv, nullptr, qkv,
                                                       BT, NQKV, EMB);
  // 2) block-local causal attention per (b, blk, h)
  attn_kernel<<<dim3(4 * 32 * HEADS), blk, 0, stream>>>(qkv, attn);
  // 3) out = attn @ W_proj + b_proj  (f16 in, f32 out)
  wmma_gemm<half_t, float, true>
      <<<dim3(EMB / 128, BT / 128), blk, 0, stream>>>(attn, Wproj, bproj, out,
                                                      BT, EMB, EMB);
}
